// NeuCom_86011015070017
// MI455X (gfx1250) — matrix operand order repaired and checked by reading, not compiled, not run on hardware
//
#include <hip/hip_runtime.h>
#include <hip/hip_bf16.h>

typedef _Float16 f16;
typedef _Float16 v16h __attribute__((ext_vector_type(16)));
typedef _Float16 v8h  __attribute__((ext_vector_type(8)));
typedef float    v8f  __attribute__((ext_vector_type(8)));

#define TT 64
#define BB 16
#define II 512
#define HH 512
#define OO 512
#define NNIN 768
#define IFP 480          // iface padded to 30 tiles of 16
#define IFR 471
#define NSLOT 256
#define WORDW 64
#define RH 4
#define EPSF 1e-6f

// ---- workspace layout (f32 units for state, then f16 fragment tiles) ----
#define M_OFF   0
#define M_CNT   (BB*NSLOT*WORDW)              // 262144
#define U_OFF   (M_OFF + M_CNT)
#define P_OFF   (U_OFF + BB*NSLOT)
#define WW_OFF  (P_OFF + BB*NSLOT)
#define L_OFF   (WW_OFF + BB*NSLOT)
#define RW_OFF  (L_OFF + BB*NSLOT*NSLOT)
#define RV_OFF  (RW_OFF + BB*RH*NSLOT)
#define PO_OFF  (RV_OFF + BB*RH*WORDW)
#define IFO_OFF (PO_OFF + BB*OO)
#define STATE_F32 (IFO_OFF + BB*IFP)          // 1359360 floats
#define FRAG_B  ((size_t)STATE_F32 * 4)       // byte offset of f16 frag region
// fragment tile = 32(K) x 16(N) f16, lane-major: 512 halves / tile
#define WHID_T  (24*32)
#define WOUT_T  (16*32)
#define WIF_T   (16*30)
#define WMO_T   (8*32)
#define WHID_H  0
#define WOUT_H  (WHID_T*512)
#define WIF_H   (WOUT_H + WOUT_T*512)
#define WMO_H   (WIF_H + WIF_T*512)

__device__ __forceinline__ float sigm(float x) { return 1.f / (1.f + expf(-x)); }
__device__ __forceinline__ float oneplus(float x) {
  float sp = (x > 20.f) ? x : log1pf(expf(x));
  return 1.f + sp;
}

// ---------------- init: zero state, M = 1e-6 ----------------
__global__ __launch_bounds__(256) void dnc_init(float* __restrict__ ws) {
  int idx = blockIdx.x * 256 + threadIdx.x;
  if (idx < STATE_F32) ws[idx] = (idx < M_CNT) ? 1e-6f : 0.f;
}

// ---------------- weight conversion to WMMA B-fragment tiles ----------------
// B-layout (v_wmma_f32_16x16x32_f16, 32x16 KxN): lane<16 holds column n0+lane,
// halves 0..7 = K k0..k0+7, 8..15 = K k0+16..k0+23; lane>=16 holds column n0+lane-16, halves 0..7 = K k0+8..k0+15, 8..15 = K k0+24..k0+31.
__global__ __launch_bounds__(512) void dnc_convert(const float* __restrict__ W,
                                                   f16* __restrict__ frag,
                                                   int NT, int realN) {
  int tile = blockIdx.x;
  int kt = tile / NT, nt = tile % NT;
  int tid = threadIdx.x;
  int lane = tid >> 4, h = tid & 15;
  int n = nt * 16 + (lane & 15);
  int k = kt * 32 + ((lane & 16) ? 8 : 0) + (h & 7) + ((h & 8) ? 16 : 0);
  float v = (n < realN) ? W[(size_t)k * realN + n] : 0.f;
  frag[(size_t)tile * 512 + tid] = (f16)v;
}

// ---------------- A-fragment from f16 LDS (16xK activations) ----------------
// A-layout (16x32 MxK): lane<16 -> row=lane, halves 0..7=K 0..7, 8..15=K 16..23;
// lane>=16 -> row=lane-16, halves 0..7=K 8..15, 8..15=K 24..31.
__device__ __forceinline__ v16h a_frag(const f16* A, int stride, int kt, int lane) {
  int m = lane & 15;
  int kb = kt * 32 + ((lane & 16) ? 8 : 0);
  v8h a0 = *(const v8h*)(A + m * stride + kb);
  v8h a1 = *(const v8h*)(A + m * stride + kb + 16);
  v16h r;
#pragma unroll
  for (int i = 0; i < 8; ++i) { r[i] = a0[i]; r[8 + i] = a1[i]; }
  return r;
}

// ---------------- dense phase: one workgroup, 16 waves, all-WMMA ----------------
// prev>=0: out[prev] = pre_out[prev] + rv * W_memout   (rv is rv_new of step prev)
// do_main: h = relu(nn_in*W_hid + b); pre_out = h*W_out; iface = h*W_iface
__global__ __launch_bounds__(512, 1)
void dnc_dense(const float* __restrict__ x, const float* __restrict__ bias,
               float* __restrict__ ws,
               const f16* __restrict__ whid, const f16* __restrict__ wout,
               const f16* __restrict__ wif, const f16* __restrict__ wmo,
               float* __restrict__ out, int t, int prev, int do_main) {
  __shared__ __align__(16) f16 Ah[BB * NNIN];   // 24KB activation staging (f16)
  __shared__ __align__(16) f16 Hh[BB * HH];     // 16KB hidden staging (f16)
  float* rv  = ws + RV_OFF;
  float* po  = ws + PO_OFF;
  float* ifc = ws + IFO_OFF;
  const int tid = threadIdx.x, wvi = tid >> 5, lane = tid & 31;
  const int mrow = lane & 15;
  const int mbase = (lane & 16) ? 8 : 0;

  if (prev >= 0) {  // fused output projection for previous step
    for (int i = tid; i < BB * RH * WORDW; i += 512) Ah[i] = (f16)rv[i];  // [16][256]
    __syncthreads();
#pragma unroll
    for (int pass = 0; pass < 2; ++pass) {
      int nt = wvi + pass * 16;
      v8f acc = {};
      for (int kt = 0; kt < 8; ++kt) {
        v16h a = a_frag(Ah, 256, kt, lane);
        const f16* bp = wmo + ((size_t)(kt * 32 + nt)) * 512 + lane * 16;
        if (kt < 7) __builtin_prefetch(wmo + ((size_t)((kt + 1) * 32 + nt)) * 512 + lane * 16, 0, 3);
        v16h b = *(const v16h*)bp;
        acc = __builtin_amdgcn_wmma_f32_16x16x32_f16(false, a, false, b, (short)0, acc, false, false);
      }
      int col = nt * 16 + mrow;
#pragma unroll
      for (int v = 0; v < 8; ++v) {
        int m = v + mbase;
        out[(size_t)prev * BB * OO + m * OO + col] = acc[v] + po[m * OO + col];
      }
    }
    __syncthreads();
  }
  if (!do_main) return;

  // stage nn_in = [x_t | rv] as f16
  for (int i = tid; i < BB * NNIN; i += 512) {
    int b_ = i / NNIN, c = i % NNIN;
    float v = (c < II) ? x[(size_t)t * BB * II + b_ * II + c] : rv[b_ * (RH * WORDW) + (c - II)];
    Ah[i] = (f16)v;
  }
  __syncthreads();

  // layer 1: h = relu(nn_in @ W_hid + b)   (24 K-tiles, 32 N-tiles)
#pragma unroll
  for (int pass = 0; pass < 2; ++pass) {
    int nt = wvi + pass * 16;
    v8f acc = {};
    for (int kt = 0; kt < 24; ++kt) {
      v16h a = a_frag(Ah, NNIN, kt, lane);
      const f16* bp = whid + ((size_t)(kt * 32 + nt)) * 512 + lane * 16;
      if (kt < 23) __builtin_prefetch(whid + ((size_t)((kt + 1) * 32 + nt)) * 512 + lane * 16, 0, 3);
      v16h b = *(const v16h*)bp;
      acc = __builtin_amdgcn_wmma_f32_16x16x32_f16(false, a, false, b, (short)0, acc, false, false);
    }
    int col = nt * 16 + mrow;
    float bv = bias[col];
#pragma unroll
    for (int v = 0; v < 8; ++v) {
      int m = v + mbase;
      float h = acc[v] + bv;
      Hh[m * HH + col] = (f16)(h > 0.f ? h : 0.f);
    }
  }
  __syncthreads();

  // pre_out = h @ W_out   (16 K-tiles, 32 N-tiles)
#pragma unroll
  for (int pass = 0; pass < 2; ++pass) {
    int nt = wvi + pass * 16;
    v8f acc = {};
    for (int kt = 0; kt < 16; ++kt) {
      v16h a = a_frag(Hh, HH, kt, lane);
      const f16* bp = wout + ((size_t)(kt * 32 + nt)) * 512 + lane * 16;
      if (kt < 15) __builtin_prefetch(wout + ((size_t)((kt + 1) * 32 + nt)) * 512 + lane * 16, 0, 3);
      v16h b = *(const v16h*)bp;
      acc = __builtin_amdgcn_wmma_f32_16x16x32_f16(false, a, false, b, (short)0, acc, false, false);
    }
    int col = nt * 16 + mrow;
#pragma unroll
    for (int v = 0; v < 8; ++v) po[(v + mbase) * OO + col] = acc[v];
  }

  // iface = h @ W_iface   (16 K-tiles, 30 N-tiles, padded to 480 cols)
#pragma unroll
  for (int pass = 0; pass < 2; ++pass) {
    int nt = wvi + pass * 16;
    if (nt >= 30) continue;
    v8f acc = {};
    for (int kt = 0; kt < 16; ++kt) {
      v16h a = a_frag(Hh, HH, kt, lane);
      const f16* bp = wif + ((size_t)(kt * 30 + nt)) * 512 + lane * 16;
      if (kt < 15) __builtin_prefetch(wif + ((size_t)((kt + 1) * 30 + nt)) * 512 + lane * 16, 0, 3);
      v16h b = *(const v16h*)bp;
      acc = __builtin_amdgcn_wmma_f32_16x16x32_f16(false, a, false, b, (short)0, acc, false, false);
    }
    int col = nt * 16 + mrow;
#pragma unroll
    for (int v = 0; v < 8; ++v) ifc[(v + mbase) * IFP + col] = acc[v];
  }
}

// ---------------- memory module: one workgroup per batch, 1 thread per slot ----------------
__global__ __launch_bounds__(256, 1)
void dnc_memory(const float* __restrict__ ifall, float* __restrict__ M,
                float* __restrict__ u, float* __restrict__ p,
                float* __restrict__ wwg, float* __restrict__ L,
                float* __restrict__ rw, float* __restrict__ rv) {
  const int b = blockIdx.x;
  const int n = threadIdx.x;
  __shared__ float rk[RH][WORDW];
  __shared__ float wk[WORDW], ev[WORDW], wvv[WORDW];
  __shared__ float rb[RH], fg[RH], rmg[RH][3];
  __shared__ float sc[4];
  __shared__ float rwl[RH][NSLOT];
  __shared__ float pl[NSLOT], ua[NSLOT], wwl[NSLOT], red[NSLOT];
  __shared__ float red4[RH][NSLOT];
  __shared__ float Ml[NSLOT][WORDW];           // 64KB: M_new staging
  __shared__ float rwn[RH][NSLOT];

  const float* ifc = ifall + (size_t)b * IFP;
  { int r = n >> 6, w = n & 63; rk[r][w] = ifc[r * WORDW + w]; }
  if (n < WORDW)            wk[n]        = ifc[260 + n];
  else if (n < 2 * WORDW)   ev[n - 64]   = sigm(ifc[325 + (n - 64)]);
  else if (n < 3 * WORDW)   wvv[n - 128] = ifc[389 + (n - 128)];
  else if (n < 192 + RH) {
    int r = n - 192;
    float a0 = ifc[459 + r * 3], a1 = ifc[460 + r * 3], a2 = ifc[461 + r * 3];
    float mx = fmaxf(a0, fmaxf(a1, a2));
    float e0 = expf(a0 - mx), e1 = expf(a1 - mx), e2 = expf(a2 - mx);
    float s = e0 + e1 + e2;
    rmg[r][0] = e0 / s; rmg[r][1] = e1 / s; rmg[r][2] = e2 / s;
  }
  if (n < RH) { rb[n] = oneplus(ifc[256 + n]); fg[n] = sigm(ifc[453 + n]); }
  if (n == 0) { sc[0] = oneplus(ifc[324]); sc[1] = sigm(ifc[457]); sc[2] = sigm(ifc[458]); }
#pragma unroll
  for (int r = 0; r < RH; ++r) rwl[r][n] = rw[((size_t)b * RH + r) * NSLOT + n];
  pl[n] = p[b * NSLOT + n];
  float u_old = u[b * NSLOT + n];
  float ww_old = wwg[b * NSLOT + n];
  __syncthreads();
  const float wbeta = sc[0], agv = sc[1], wgv = sc[2];

  // usage / retention
  float ret = 1.f;
#pragma unroll
  for (int r = 0; r < RH; ++r) ret *= (1.f - fg[r] * rwl[r][n]);
  float un = (u_old + ww_old - u_old * ww_old) * ret;
  ua[n] = un; u[b * NSLOT + n] = un;
  __syncthreads();

  // allocation: stable rank + product scan (cumprod of sorted usage)
  int rank = 0;
  for (int m = 0; m < NSLOT; ++m) {
    float um = ua[m];
    rank += (um < un) || (um == un && m < n);
  }
  red[rank] = un;
  __syncthreads();
  for (int off = 1; off < NSLOT; off <<= 1) {
    float tp = (n >= off) ? red[n - off] : 1.f;
    float mine = red[n];
    __syncthreads();
    red[n] = mine * tp;
    __syncthreads();
  }
  float cp = (rank > 0) ? red[rank - 1] : 1.f;
  float aalloc = (1.f - un) * cp;
  __syncthreads();

  // content write weights on M_old
  const float* Mr = M + ((size_t)b * NSLOT + n) * WORDW;
  float dot = 0.f, mn2 = 0.f, kn2 = 0.f;
  for (int w = 0; w < WORDW; ++w) {
    float mv = Mr[w], kv = wk[w];
    dot += mv * kv; mn2 += mv * mv; kn2 += kv * kv;
  }
  float logit = wbeta * dot / ((sqrtf(mn2) + EPSF) * (sqrtf(kn2) + EPSF));
  red[n] = logit; __syncthreads();
  for (int off = 128; off > 0; off >>= 1) { if (n < off) red[n] = fmaxf(red[n], red[n + off]); __syncthreads(); }
  float mx = red[0]; __syncthreads();
  float e = expf(logit - mx);
  red[n] = e; __syncthreads();
  for (int off = 128; off > 0; off >>= 1) { if (n < off) red[n] += red[n + off]; __syncthreads(); }
  float cw = e / red[0];
  __syncthreads();

  // write weighting + sum
  float wwn = wgv * (agv * aalloc + (1.f - agv) * cw);
  wwl[n] = wwn; wwg[b * NSLOT + n] = wwn;
  red[n] = wwn; __syncthreads();
  for (int off = 128; off > 0; off >>= 1) { if (n < off) red[n] += red[n + off]; __syncthreads(); }
  float sww = red[0];
  __syncthreads();

  // memory update (stage M_new in LDS)
  float* Mw = M + ((size_t)b * NSLOT + n) * WORDW;
  for (int w = 0; w < WORDW; ++w) {
    float m0 = Mw[w];
    float mnew = m0 * (1.f - wwn * ev[w]) + wwn * wvv[w];
    Mw[w] = mnew; Ml[n][w] = mnew;
  }

  // bwd link read from L_old (column pass, coalesced) BEFORE overwriting L
  float ba[RH] = {0.f, 0.f, 0.f, 0.f};
  const float* Lb = L + (size_t)b * NSLOT * NSLOT;
  for (int m = 0; m < NSLOT; ++m) {
    float lv = Lb[(size_t)m * NSLOT + n];
    float ln = (1.f - wwl[m] - wwn) * lv + wwl[m] * pl[n];
    ln = (m == n) ? 0.f : ln;
#pragma unroll
    for (int r = 0; r < RH; ++r) ba[r] += ln * rwl[r][m];
  }
  __syncthreads();

  // L_new row pass (write) fused with fwd link read
  float fa[RH] = {0.f, 0.f, 0.f, 0.f};
  float* Lr = L + ((size_t)b * NSLOT + n) * NSLOT;
  for (int j = 0; j < NSLOT; ++j) {
    float lv = Lr[j];
    float ln = (1.f - wwn - wwl[j]) * lv + wwn * pl[j];
    ln = (j == n) ? 0.f : ln;
    Lr[j] = ln;
#pragma unroll
    for (int r = 0; r < RH; ++r) fa[r] += ln * rwl[r][j];
  }
  p[b * NSLOT + n] = (1.f - sww) * pl[n] + wwn;
  __syncthreads();

  // read content weights on M_new (4 heads, parallel softmaxes)
  float lg[RH];
#pragma unroll
  for (int r = 0; r < RH; ++r) {
    float d = 0.f, m2 = 0.f, k2 = 0.f;
    for (int w = 0; w < WORDW; ++w) {
      float mv = Ml[n][w], kv = rk[r][w];
      d += mv * kv; m2 += mv * mv; k2 += kv * kv;
    }
    lg[r] = rb[r] * d / ((sqrtf(m2) + EPSF) * (sqrtf(k2) + EPSF));
    red4[r][n] = lg[r];
  }
  __syncthreads();
  for (int off = 128; off > 0; off >>= 1) {
    if (n < off)
#pragma unroll
      for (int r = 0; r < RH; ++r) red4[r][n] = fmaxf(red4[r][n], red4[r][n + off]);
    __syncthreads();
  }
  float mx4[RH];
#pragma unroll
  for (int r = 0; r < RH; ++r) mx4[r] = red4[r][0];
  __syncthreads();
#pragma unroll
  for (int r = 0; r < RH; ++r) { lg[r] = expf(lg[r] - mx4[r]); red4[r][n] = lg[r]; }
  __syncthreads();
  for (int off = 128; off > 0; off >>= 1) {
    if (n < off)
#pragma unroll
      for (int r = 0; r < RH; ++r) red4[r][n] += red4[r][n + off];
    __syncthreads();
  }

  // rw_new = pi0*bwd + pi1*rc + pi2*fwd
#pragma unroll
  for (int r = 0; r < RH; ++r) {
    float rc = lg[r] / red4[r][0];
    float rn = rmg[r][0] * ba[r] + rmg[r][1] * rc + rmg[r][2] * fa[r];
    rwn[r][n] = rn;
    rw[((size_t)b * RH + r) * NSLOT + n] = rn;
  }
  __syncthreads();

  // rv = M_new^T rw_new : thread = (head, word)
  {
    int r = n >> 6, w = n & 63;
    float acc = 0.f;
    for (int m = 0; m < NSLOT; ++m) acc += Ml[m][w] * rwn[r][m];
    rv[((size_t)b * RH + r) * WORDW + w] = acc;
  }
}

// ---------------- host launch ----------------
extern "C" void kernel_launch(void* const* d_in, const int* in_sizes, int n_in,
                              void* d_out, int out_size, void* d_ws, size_t ws_size,
                              hipStream_t stream) {
  (void)in_sizes; (void)n_in; (void)out_size; (void)ws_size;
  const float* x    = (const float*)d_in[0];
  const float* Whid = (const float*)d_in[1];
  const float* bh   = (const float*)d_in[2];
  const float* Wif  = (const float*)d_in[3];
  const float* Wout = (const float*)d_in[4];
  const float* Wmo  = (const float*)d_in[5];
  float* out = (float*)d_out;
  float* ws  = (float*)d_ws;
  f16* frag = (f16*)((char*)d_ws + FRAG_B);

  dnc_init<<<(STATE_F32 + 255) / 256, 256, 0, stream>>>(ws);
  dnc_convert<<<WHID_T, 512, 0, stream>>>(Whid, frag + WHID_H, 32, 512);
  dnc_convert<<<WOUT_T, 512, 0, stream>>>(Wout, frag + WOUT_H, 32, 512);
  dnc_convert<<<WIF_T,  512, 0, stream>>>(Wif,  frag + WIF_H,  30, 471);
  dnc_convert<<<WMO_T,  512, 0, stream>>>(Wmo,  frag + WMO_H,  32, 512);

  for (int t = 0; t < TT; ++t) {
    dnc_dense<<<1, 512, 0, stream>>>(x, bh, ws,
                                     frag + WHID_H, frag + WOUT_H, frag + WIF_H, frag + WMO_H,
                                     out, t, t - 1, 1);
    dnc_memory<<<BB, 256, 0, stream>>>(ws + IFO_OFF, ws + M_OFF, ws + U_OFF, ws + P_OFF,
                                       ws + WW_OFF, ws + L_OFF, ws + RW_OFF, ws + RV_OFF);
  }
  // epilogue: emit out[T-1] from the final rv
  dnc_dense<<<1, 512, 0, stream>>>(x, bh, ws,
                                   frag + WHID_H, frag + WOUT_H, frag + WIF_H, frag + WMO_H,
                                   out, 0, TT - 1, 0);
}
